// AxialAttention_34694745817832
// MI455X (gfx1250) — matrix operand order repaired and checked by reading, not compiled, not run on hardware
//
#include <hip/hip_runtime.h>
#include <stdint.h>

typedef __bf16 bf16;
typedef __attribute__((ext_vector_type(16))) __bf16 v16bf;
typedef __attribute__((ext_vector_type(8)))  __bf16 v8bf;
typedef __attribute__((ext_vector_type(8)))  float  v8f;
typedef __attribute__((ext_vector_type(4)))  unsigned int v4u;
typedef __attribute__((ext_vector_type(8)))  int v8i;
typedef __attribute__((ext_vector_type(4)))  int v4i;

#if defined(__gfx1250__) && defined(__has_builtin)
#  if __has_builtin(__builtin_amdgcn_tensor_load_to_lds) && __has_builtin(__builtin_amdgcn_s_wait_tensorcnt)
#    define USE_TDM 1
#  endif
#endif
#ifndef USE_TDM
#  define USE_TDM 0
#endif

#define N_BATCH 16
#define CIN     512
#define HH      64
#define WW      64
#define HEADS   8

// ---- workspace byte offsets ----
#define WS_W    0u                      // w_qkv as bf16: 1024*512*2 = 1 MB
#define WS_EMB  1048576u                // embT[i][cc][j] bf16: 64*128*64*2 = 1 MB
#define WS_SQ   2097152u                // bnq folded scale (1024 f32)
#define WS_BQ   (WS_SQ + 4096u)        // bnq folded bias
#define WS_SIMF (WS_BQ + 4096u)        // per-head a0,a1,a2,beta (8*4 f32)
#define WS_S0   (WS_SIMF + 128u)       // bno scale for sv   (512 f32)
#define WS_S1   (WS_S0 + 2048u)        // bno scale for sve  (512 f32)
#define WS_BS   (WS_S1 + 2048u)        // bno summed bias    (512 f32)

// ---- LDS layout (dynamic) ----
#define SLAB_OFF   0                   // emb slab: 64 rows x pitch 72 bf16 (144B rows, TDM-paddable)
#define SLAB_PITCH 72
#define XBT_OFF    16384               // x slab: 64 rows x pitch 520 bf16 (dead after GEMM)
#define XBT_PITCH  520
#define VS_OFF     16384               // per-wave: vA (64x72 bf16 = 9216) + sim (64x65 f32 = 16640)
#define VS_STRIDE  25856
#define VA_PITCH   72
#define SIM_PITCH  65
#define QK_OFF     223232              // per-wave: qA (64x40 bf16) + kT (64x40 bf16), disjoint from xbt
#define QK_STRIDE  10240
#define QA_PITCH   40
#define LDS_BYTES  305152              // 16384 + 8*25856 + 8*10240  (<= 320KB WGP LDS)

__device__ __forceinline__ v8f vzero() {
  v8f z = {0.f,0.f,0.f,0.f,0.f,0.f,0.f,0.f};
  return z;
}
// A-operand: per-lane K {0..15} (lane<16) / {16..31} (lane>=16);
// caller passes p already offset by the lane's first K.
__device__ __forceinline__ v16bf load16_split(const bf16* p) {
  v16bf r;
  ((uint4*)&r)[0] = *(const uint4*)(p);
  ((uint4*)&r)[1] = *(const uint4*)(p + 8);
  return r;
}
// B-operand: 16 contiguous K values per lane.
__device__ __forceinline__ v16bf load16_contig(const bf16* p) {
  v16bf r;
  ((uint4*)&r)[0] = *(const uint4*)(p);
  ((uint4*)&r)[1] = *(const uint4*)(p + 8);
  return r;
}

#if USE_TDM
// TDM: 64x64 bf16 2D tile, contiguous in memory, LDS-padded from 128B rows to 144B pitch.
// 6-arg builtin form (clang-23 / therock-10.0 headers).
__device__ __forceinline__ void tdm_load_slab64(unsigned long long gaddr, unsigned lds_addr) {
  v4u g0;
  g0[0] = 1u;                                            // count=1, user descriptor
  g0[1] = lds_addr;                                      // LDS byte address
  g0[2] = (unsigned)(gaddr & 0xffffffffull);             // global_addr[31:0]
  g0[3] = (unsigned)((gaddr >> 32) & 0x01ffffffull) | (2u << 30);  // addr[56:32] | type=2
  v8i g1;
  g1[0] = (int)((1u<<16)      // data_size = 1 (2 bytes)
              | (1u<<20)      // pad_enable
              | (4u<<22)      // pad_interval: every 32 dwords (128B row)
              | (3u<<25));    // pad_amount: 4 dwords (16B) -> 144B LDS pitch
  g1[1] = (int)(64u << 16);   // tensor_dim0 = 64
  g1[2] = (int)(64u << 16);   // tensor_dim1 = 64
  g1[3] = (int)(64u << 16);   // tile_dim0 = 64
  g1[4] = 64;                 // tile_dim1 = 64, tile_dim2 = 0
  g1[5] = 64;                 // tensor_dim0_stride = 64
  g1[6] = 0;
  g1[7] = 0;
  v4i gz4; gz4[0]=0; gz4[1]=0; gz4[2]=0; gz4[3]=0;       // D# groups 2,3: unused (<=2D tensor)
  v8i gz8;
#pragma unroll
  for (int e = 0; e < 8; ++e) gz8[e] = 0;                // extended group: zero (no cluster/iterate)
  __builtin_amdgcn_tensor_load_to_lds(g0, g1, gz4, gz4, gz8, 0);
}
#endif

// =============== prep: bf16 tables + folded BN coefficients ===============
__global__ __launch_bounds__(256) void prep_kernel(
    const float* __restrict__ wq, const float* __restrict__ rel,
    const float* __restrict__ bnqg, const float* __restrict__ bnqb,
    const float* __restrict__ bnqm, const float* __restrict__ bnqv,
    const float* __restrict__ bnsg, const float* __restrict__ bnsb,
    const float* __restrict__ bnsm, const float* __restrict__ bnsv,
    const float* __restrict__ bnog, const float* __restrict__ bnob,
    const float* __restrict__ bnom, const float* __restrict__ bnov,
    unsigned char* __restrict__ ws)
{
  const int T = blockIdx.x * 256 + threadIdx.x;   // grid = 256 blocks -> 65536 threads
  bf16* wb  = (bf16*)(ws + WS_W);
  bf16* emb = (bf16*)(ws + WS_EMB);
  {  // w_qkv fp32 -> bf16 (8 elems/thread, exactly 1024*512)
    const int idx = T * 8;
    v8bf t;
#pragma unroll
    for (int e = 0; e < 8; ++e) t[e] = (bf16)wq[idx + e];
    *(v8bf*)(wb + idx) = t;
  }
  {  // embT[i][cc][j] = relative[cc][i-j+63]  (gather; 8 elems/thread)
    const int f0 = T * 8;
    const int i  = f0 >> 13;          // /(128*64)
    const int cc = (f0 >> 6) & 127;
    const int j0 = f0 & 63;
    v8bf t;
#pragma unroll
    for (int e = 0; e < 8; ++e) t[e] = (bf16)rel[cc * 127 + (i - (j0 + e) + 63)];
    *(v8bf*)(emb + f0) = t;
  }
  if (T < 1024) {                       // fold bnq
    float* sq = (float*)(ws + WS_SQ);
    float* bq = (float*)(ws + WS_BQ);
    const float s = bnqg[T] * rsqrtf(bnqv[T] + 1e-5f);
    sq[T] = s;
    bq[T] = bnqb[T] - bnqm[T] * s;
  } else if (T < 1032) {                // fold bns + 3-way sum: sim = a0*qk+a1*qr+a2*kr+beta
    const int gh = T - 1024;
    float* sf = (float*)(ws + WS_SIMF);
    float beta = 0.f;
#pragma unroll
    for (int t3 = 0; t3 < 3; ++t3) {
      const int ch = t3 * 8 + gh;
      const float s = bnsg[ch] * rsqrtf(bnsv[ch] + 1e-5f);
      sf[gh * 4 + t3] = s;
      beta += bnsb[ch] - bnsm[ch] * s;
    }
    sf[gh * 4 + 3] = beta;
  } else if (T < 1544) {                // fold bno + pair-sum: out = s0*sv + s1*sve + bsum
    const int c = T - 1032;
    float* s0 = (float*)(ws + WS_S0);
    float* s1 = (float*)(ws + WS_S1);
    float* bs = (float*)(ws + WS_BS);
    const int ch0 = 2 * c, ch1 = ch0 + 1;
    const float t0 = bnog[ch0] * rsqrtf(bnov[ch0] + 1e-5f);
    const float t1 = bnog[ch1] * rsqrtf(bnov[ch1] + 1e-5f);
    s0[c] = t0; s1[c] = t1;
    bs[c] = (bnob[ch0] - bnom[ch0] * t0) + (bnob[ch1] - bnom[ch1] * t1);
  }
}

// =============== fused axial attention: 1 block per (n,w), 1 wave per head ===============
__global__ __launch_bounds__(256) void axial_fused(
    const float* __restrict__ x,
    const unsigned char* __restrict__ ws,
    float* __restrict__ out)
{
  extern __shared__ char smem[];
  const int bw   = blockIdx.x;
  const int n    = bw >> 6;
  const int w    = bw & 63;
  const int tid  = threadIdx.x;
  const int wave = tid >> 5;
  const int lane = tid & 31;
  const int lhalf = (lane < 16) ? 0 : 1;
  const int l15   = lane & 15;
  const int g = wave;   // head

  const bf16*  wb   = (const bf16*)(ws + WS_W);
  const bf16*  emb  = (const bf16*)(ws + WS_EMB);
  const float* sq   = (const float*)(ws + WS_SQ);
  const float* bq   = (const float*)(ws + WS_BQ);
  const float* simf = (const float*)(ws + WS_SIMF);
  const float* s0t  = (const float*)(ws + WS_S0);
  const float* s1t  = (const float*)(ws + WS_S1);
  const float* bst  = (const float*)(ws + WS_BS);

  bf16*  slab = (bf16*)(smem + SLAB_OFF);
  bf16*  xbt  = (bf16*)(smem + XBT_OFF);
  bf16*  vA   = (bf16*)(smem + VS_OFF + wave * VS_STRIDE);            // [c][h] A-layout for sv
  float* sim  = (float*)(smem + VS_OFF + wave * VS_STRIDE + 9216);    // [i][j] pitch 65
  bf16*  qA   = (bf16*)(smem + QK_OFF + wave * QK_STRIDE);            // [h][u] A-layout for qk
  bf16*  kT   = (bf16*)(smem + QK_OFF + wave * QK_STRIDE + 5120);     // [h][u] B-layout for qk
  bf16*  sveW = (bf16*)(smem + VS_OFF + wave * VS_STRIDE);            // overlays vA after sv GEMM

  // ---- Phase A: x[n,:,:,w] -> xbt[h][c] (bf16, B-layout for QKV GEMM) ----
  {
    const int h  = tid & 63;
    const int cg = tid >> 6;
    const float* xp = x + (size_t)n * CIN * HH * WW + (size_t)h * WW + w;
#pragma unroll 1
    for (int c = cg; c < CIN; c += 4)
      xbt[h * XBT_PITCH + c] = (bf16)xp[(size_t)c * HH * WW];
  }
  __syncthreads();

  // ---- Phase B: QKV GEMM, wave g computes rows o = g*128 .. g*128+127, K = 512 ----
  auto gemm_half = [&](int half, v8f (&acc)[4][4]) {
#pragma unroll
    for (int a = 0; a < 4; ++a)
#pragma unroll
      for (int b = 0; b < 4; ++b) acc[a][b] = vzero();
#pragma unroll 1
    for (int kc = 0; kc < 16; ++kc) {
      const int cbase = kc * 32;
      v16bf Bt[4];
#pragma unroll
      for (int ht = 0; ht < 4; ++ht)
        Bt[ht] = load16_contig(xbt + (ht * 16 + l15) * XBT_PITCH + cbase + lhalf * 16);
#pragma unroll
      for (int ot = 0; ot < 4; ++ot) {
        const int o = g * 128 + half * 64 + ot * 16 + l15;
        v16bf At = load16_split(wb + (size_t)o * CIN + cbase + lhalf * 16);
#pragma unroll
        for (int ht = 0; ht < 4; ++ht)
          acc[ot][ht] = __builtin_amdgcn_wmma_f32_16x16x32_bf16(
              false, At, false, Bt[ht], (short)0, acc[ot][ht], false, false);
      }
    }
  };

  v8f accQ[4][4], accV[4][4];
  gemm_half(0, accQ);          // u = 0..63  -> q (0..31), k (32..63)
  // stage q,k with BN fold (region disjoint from xbt: no barrier needed)
#pragma unroll
  for (int ot = 0; ot < 4; ++ot)
#pragma unroll
    for (int ht = 0; ht < 4; ++ht) {
      const int hcol = ht * 16 + l15;
#pragma unroll
      for (int r = 0; r < 8; ++r) {
        const int u = ot * 16 + r + lhalf * 8;
        const int o = g * 128 + u;
        const float val = accQ[ot][ht][r] * sq[o] + bq[o];
        if (u < 32) qA[hcol * QA_PITCH + u]        = (bf16)val;   // transposed: [h][u]
        else        kT[hcol * QA_PITCH + (u - 32)] = (bf16)val;   // transposed: [h][u]
      }
    }
  gemm_half(1, accV);          // u = 64..127 -> v
  __syncthreads();             // all waves done reading xbt (vA/sim overlay it)

  // stage v with BN fold: vA[c][h]
#pragma unroll
  for (int ot = 0; ot < 4; ++ot)
#pragma unroll
    for (int ht = 0; ht < 4; ++ht) {
      const int hcol = ht * 16 + l15;
#pragma unroll
      for (int r = 0; r < 8; ++r) {
        const int c = ot * 16 + r + lhalf * 8;
        const int o = g * 128 + 64 + c;
        const float val = accV[ot][ht][r] * sq[o] + bq[o];
        vA[c * VA_PITCH + hcol] = (bf16)val;
      }
    }

  // ---- qk = q^T k via WMMA; dump sim = a0*qk + beta ----
  const float a0 = simf[g * 4 + 0], a1 = simf[g * 4 + 1];
  const float a2 = simf[g * 4 + 2], beta = simf[g * 4 + 3];
#pragma unroll
  for (int it = 0; it < 4; ++it) {
    v16bf Aq = load16_split(qA + (it * 16 + l15) * QA_PITCH + lhalf * 16);
#pragma unroll
    for (int jt = 0; jt < 4; ++jt) {
      const int j = jt * 16 + l15;
      v16bf Bk = load16_contig(kT + j * QA_PITCH + lhalf * 16);
      v8f d = __builtin_amdgcn_wmma_f32_16x16x32_bf16(
          false, Aq, false, Bk, (short)0, vzero(), false, false);
#pragma unroll
      for (int r = 0; r < 8; ++r)
        sim[(it * 16 + r + lhalf * 8) * SIM_PITCH + j] = a0 * d[r] + beta;
    }
  }

  // ---- qr/kr: lockstep over slab s; slab = embT[s][0:64][:] (q rows 0..31, k rows 32..63) ----
#pragma unroll 1
  for (int s = 0; s < 64; ++s) {
    __syncthreads();
#if USE_TDM
    if (wave == 0) {
      tdm_load_slab64((unsigned long long)(uintptr_t)(emb + (size_t)s * 8192),
                      (unsigned)(uintptr_t)(void*)slab);
      __builtin_amdgcn_s_wait_tensorcnt(0);
    }
#else
    {
      const int rr = tid >> 2, qo = (tid & 3) * 16;
      const uint4* sp = (const uint4*)(emb + (size_t)s * 8192 + rr * 64 + qo);
      uint4* dp = (uint4*)(slab + rr * SLAB_PITCH + qo);
      dp[0] = sp[0]; dp[1] = sp[1];
    }
#endif
    __syncthreads();
    float q0 = 0.f, q1 = 0.f, k0 = 0.f, k1 = 0.f;
#pragma unroll 4
    for (int c = 0; c < 32; ++c) {
      const float qc = (float)qA[s * QA_PITCH + c];   // q[c][i=s] (broadcast)
      q0 += qc * (float)slab[c * SLAB_PITCH + lane];
      q1 += qc * (float)slab[c * SLAB_PITCH + lane + 32];
      const float kc = (float)kT[s * QA_PITCH + c];   // k[c][j=s] (broadcast)
      k0 += kc * (float)slab[(32 + c) * SLAB_PITCH + lane];
      k1 += kc * (float)slab[(32 + c) * SLAB_PITCH + lane + 32];
    }
    sim[s * SIM_PITCH + lane]        += a1 * q0;   // qr row s
    sim[s * SIM_PITCH + lane + 32]   += a1 * q1;
    sim[lane * SIM_PITCH + s]        += a2 * k0;   // kr column s (pitch 65 -> conflict-free)
    sim[(lane + 32) * SIM_PITCH + s] += a2 * k1;
  }

  // ---- softmax over j (wave-local, shfl reductions) ----
#pragma unroll 1
  for (int i = 0; i < 64; ++i) {
    float v0 = sim[i * SIM_PITCH + lane];
    float v1 = sim[i * SIM_PITCH + lane + 32];
    float m = fmaxf(v0, v1);
#pragma unroll
    for (int off = 16; off > 0; off >>= 1) m = fmaxf(m, __shfl_xor(m, off, 32));
    const float e0 = __expf(v0 - m), e1 = __expf(v1 - m);
    float ssum = e0 + e1;
#pragma unroll
    for (int off = 16; off > 0; off >>= 1) ssum += __shfl_xor(ssum, off, 32);
    const float inv = 1.0f / ssum;
    sim[i * SIM_PITCH + lane]      = e0 * inv;
    sim[i * SIM_PITCH + lane + 32] = e1 * inv;
  }

  // ---- sv[c][i] = sum_j v[c][j] * sim[i][j] via WMMA (A = vA, B = sim rows, K = j) ----
  v8f sv[4][4];
#pragma unroll
  for (int a = 0; a < 4; ++a)
#pragma unroll
    for (int b = 0; b < 4; ++b) sv[a][b] = vzero();
#pragma unroll
  for (int jc = 0; jc < 2; ++jc) {
    const int jb = jc * 32;
    v16bf Bs[4];
#pragma unroll
    for (int it = 0; it < 4; ++it) {
      const float* sp = sim + (it * 16 + l15) * SIM_PITCH + jb + lhalf * 16;
      v16bf t;
#pragma unroll
      for (int e = 0; e < 16; ++e) t[e] = (bf16)sp[e];
      Bs[it] = t;
    }
#pragma unroll
    for (int ct = 0; ct < 4; ++ct) {
      v16bf Av = load16_split(vA + (ct * 16 + l15) * VA_PITCH + jb + lhalf * 16);
#pragma unroll
      for (int it = 0; it < 4; ++it)
        sv[ct][it] = __builtin_amdgcn_wmma_f32_16x16x32_bf16(
            false, Av, false, Bs[it], (short)0, sv[ct][it], false, false);
    }
  }

  // ---- sve: lockstep over slab s; slab = embT[s][64:128][:] ; sveW overlays dead vA ----
#pragma unroll 1
  for (int s = 0; s < 64; ++s) {
    __syncthreads();
#if USE_TDM
    if (wave == 0) {
      tdm_load_slab64((unsigned long long)(uintptr_t)(emb + (size_t)s * 8192 + 4096),
                      (unsigned)(uintptr_t)(void*)slab);
      __builtin_amdgcn_s_wait_tensorcnt(0);
    }
#else
    {
      const int rr = tid >> 2, qo = (tid & 3) * 16;
      const uint4* sp = (const uint4*)(emb + (size_t)s * 8192 + 4096 + rr * 64 + qo);
      uint4* dp = (uint4*)(slab + rr * SLAB_PITCH + qo);
      dp[0] = sp[0]; dp[1] = sp[1];
    }
#endif
    __syncthreads();
    float acc0 = 0.f, acc1 = 0.f;   // c = lane, lane+32
#pragma unroll 4
    for (int j = 0; j < 64; ++j) {
      const float sj = sim[s * SIM_PITCH + j];                 // broadcast
      acc0 += sj * (float)slab[lane * SLAB_PITCH + j];
      acc1 += sj * (float)slab[(lane + 32) * SLAB_PITCH + j];
    }
    sveW[lane * 66 + s]        = (bf16)acc0;
    sveW[(lane + 32) * 66 + s] = (bf16)acc1;
  }

  // ---- epilogue: out[n, g*64+c, i, w] = s0*sv + s1*sve + bsum ----
#pragma unroll
  for (int ct = 0; ct < 4; ++ct)
#pragma unroll
    for (int it = 0; it < 4; ++it) {
      const int i = it * 16 + l15;
#pragma unroll
      for (int r = 0; r < 8; ++r) {
        const int c    = ct * 16 + r + lhalf * 8;
        const int c512 = g * 64 + c;
        const float val = s0t[c512] * sv[ct][it][r]
                        + s1t[c512] * (float)sveW[c * 66 + i]
                        + bst[c512];
        out[(((size_t)n * CIN + c512) * HH + i) * WW + w] = val;
      }
    }
}

extern "C" void kernel_launch(void* const* d_in, const int* in_sizes, int n_in,
                              void* d_out, int out_size, void* d_ws, size_t ws_size,
                              hipStream_t stream) {
  (void)in_sizes; (void)n_in; (void)out_size; (void)ws_size;
  const float* x    = (const float*)d_in[0];
  const float* wq   = (const float*)d_in[1];
  const float* rel  = (const float*)d_in[2];
  const float* bnqg = (const float*)d_in[3];
  const float* bnqb = (const float*)d_in[4];
  const float* bnqm = (const float*)d_in[5];
  const float* bnqv = (const float*)d_in[6];
  const float* bnsg = (const float*)d_in[7];
  const float* bnsb = (const float*)d_in[8];
  const float* bnsm = (const float*)d_in[9];
  const float* bnsv = (const float*)d_in[10];
  const float* bnog = (const float*)d_in[11];
  const float* bnob = (const float*)d_in[12];
  const float* bnom = (const float*)d_in[13];
  const float* bnov = (const float*)d_in[14];
  unsigned char* ws = (unsigned char*)d_ws;

  prep_kernel<<<256, 256, 0, stream>>>(wq, rel, bnqg, bnqb, bnqm, bnqv,
                                       bnsg, bnsb, bnsm, bnsv,
                                       bnog, bnob, bnom, bnov, ws);
  (void)hipFuncSetAttribute(reinterpret_cast<const void*>(axial_fused),
                            hipFuncAttributeMaxDynamicSharedMemorySize, LDS_BYTES);
  axial_fused<<<1024, 256, LDS_BYTES, stream>>>(x, ws, (float*)d_out);
}
